// QueryMixer_86998857548012
// MI455X (gfx1250) — matrix operand order repaired and checked by reading, not compiled, not run on hardware
//
#include <hip/hip_runtime.h>
#include <stdint.h>

typedef __attribute__((ext_vector_type(16))) __bf16         v16bf;
typedef __attribute__((ext_vector_type(16))) unsigned short v16u;
typedef __attribute__((ext_vector_type(8)))  float          v8f;
typedef __attribute__((ext_vector_type(4)))  float          v4f;
typedef __attribute__((ext_vector_type(4)))  unsigned int   v4u;

#define B_ 8192
#define N_ 16
#define D_ 256
#define H_ 512

__device__ __forceinline__ unsigned short f2bf(float f) {
  unsigned u = __float_as_uint(f);
  unsigned r = u + 0x7FFFu + ((u >> 16) & 1u);   // round-to-nearest-even
  return (unsigned short)(r >> 16);
}

// ---------------------------------------------------------------------------
// k0: swizzle f32 weights into bf16 WMMA B-operand layout.
// B operand (bf16, 32x16): lane<16 -> col=lane, K=0..7,16..23 ; lane>=16 -> col=lane-16, K=8..15,24..31
// w1B/w3B records: [n][ht(32)][kc(8)][lane(32)] x 16 bf16
// w2B   records:  [n][dt(16)][kc2(16)][lane(32)] x 16 bf16
// ---------------------------------------------------------------------------
__global__ __launch_bounds__(256) void k0_wswz(
    const float* __restrict__ w1, const float* __restrict__ w3,
    const float* __restrict__ w2,
    unsigned short* __restrict__ w1B, unsigned short* __restrict__ w3B,
    unsigned short* __restrict__ w2B) {
  int tid = blockIdx.x * 256 + threadIdx.x;
  int mat = tid >> 17;          // 0:w1 1:w3 2:w2  (131072 threads each)
  int r   = tid & 131071;
  int lane = r & 31;
  unsigned short tmp[16];
  if (mat < 2) {
    int kc = (r >> 5) & 7, ht = (r >> 8) & 31, n = (r >> 13) & 15;
    const float* s = mat ? w3 : w1;
    unsigned short* dp = mat ? w3B : w1B;
    int colh  = ht * 16 + (lane & 15);
    int dbase = kc * 32 + ((lane & 16) ? 8 : 0);
#pragma unroll
    for (int e = 0; e < 16; e++)
      tmp[e] = f2bf(s[((size_t)(n * 256 + dbase + e + (e & 8))) * 512 + colh]);
    size_t off = ((size_t)(((n * 32 + ht) * 8 + kc) * 32 + lane)) * 16;
    v4u q0, q1;
#pragma unroll
    for (int i = 0; i < 4; i++) {
      q0[i] = (unsigned)tmp[2 * i]     | ((unsigned)tmp[2 * i + 1] << 16);
      q1[i] = (unsigned)tmp[8 + 2 * i] | ((unsigned)tmp[9 + 2 * i] << 16);
    }
    *(v4u*)(dp + off)     = q0;
    *(v4u*)(dp + off + 8) = q1;
  } else {
    int kc2 = (r >> 5) & 15, dt = (r >> 9) & 15, n = (r >> 13) & 15;
    int cold  = dt * 16 + (lane & 15);
    int hbase = kc2 * 32 + ((lane & 16) ? 8 : 0);
#pragma unroll
    for (int e = 0; e < 16; e++)
      tmp[e] = f2bf(w2[((size_t)(n * 512 + hbase + e + (e & 8))) * 256 + cold]);
    size_t off = ((size_t)(((n * 16 + dt) * 16 + kc2) * 32 + lane)) * 16;
    v4u q0, q1;
#pragma unroll
    for (int i = 0; i < 4; i++) {
      q0[i] = (unsigned)tmp[2 * i]     | ((unsigned)tmp[2 * i + 1] << 16);
      q1[i] = (unsigned)tmp[8 + 2 * i] | ((unsigned)tmp[9 + 2 * i] << 16);
    }
    *(v4u*)(w2B + off)     = q0;
    *(v4u*)(w2B + off + 8) = q1;
  }
}

// ---------------------------------------------------------------------------
// k1: P = head_mix(rmsnorm1(x)) + x ; pn = rmsnorm2(P)
//   Pbuf: natural f32 [b][n][d]
//   pnA : bf16 A-operand layout, records [n][rb(512)][kc(8)][lane(32)] x 16
// one block per b; wave j handles row (b, j); lane l holds d = 8l..8l+7
// ---------------------------------------------------------------------------
__global__ __launch_bounds__(512) void k1_mix(
    const float* __restrict__ x, const float* __restrict__ n1w,
    const float* __restrict__ n2w,
    float* __restrict__ Pbuf, unsigned short* __restrict__ pnA) {
  __shared__ float XN[16][256];
  int b = blockIdx.x;
  int j = threadIdx.x >> 5;
  int l = threadIdx.x & 31;
  size_t rowoff = (((size_t)b * 16) + j) * 256 + l * 8;

  float xv[8];
  v4f x0 = *(const v4f*)(x + rowoff);
  v4f x1 = *(const v4f*)(x + rowoff + 4);
#pragma unroll
  for (int i = 0; i < 4; i++) { xv[i] = x0[i]; xv[4 + i] = x1[i]; }

  float s = 0.f;
#pragma unroll
  for (int i = 0; i < 8; i++) s += xv[i] * xv[i];
#pragma unroll
  for (int m = 16; m > 0; m >>= 1) s += __shfl_xor(s, m, 32);
  float sc = rsqrtf(s * (1.0f / 256.0f) + 1e-6f);

  int d0 = l * 8;
#pragma unroll
  for (int i = 0; i < 8; i++) XN[j][d0 + i] = xv[i] * sc * n1w[d0 + i];
  __syncthreads();

  // P[b,j,d] = xn[b, d>>4, j*16 + (d&15)] + x[b,j,d]
  int nsrc = l >> 1;
  int cbase = j * 16 + (l & 1) * 8;
  float pv[8];
#pragma unroll
  for (int i = 0; i < 8; i++) pv[i] = XN[nsrc][cbase + i] + xv[i];

  v4f p0, p1;
#pragma unroll
  for (int i = 0; i < 4; i++) { p0[i] = pv[i]; p1[i] = pv[4 + i]; }
  *(v4f*)(Pbuf + rowoff)     = p0;
  *(v4f*)(Pbuf + rowoff + 4) = p1;

  float s2 = 0.f;
#pragma unroll
  for (int i = 0; i < 8; i++) s2 += pv[i] * pv[i];
#pragma unroll
  for (int m = 16; m > 0; m >>= 1) s2 += __shfl_xor(s2, m, 32);
  float sc2 = rsqrtf(s2 * (1.0f / 256.0f) + 1e-6f);

  unsigned short pb[8];
#pragma unroll
  for (int i = 0; i < 8; i++) pb[i] = f2bf(pv[i] * sc2 * n2w[d0 + i]);
  v4u q;
#pragma unroll
  for (int i = 0; i < 4; i++)
    q[i] = (unsigned)pb[2 * i] | ((unsigned)pb[2 * i + 1] << 16);

  // scatter 16B group into A-operand layout
  int rb = b >> 4, m = b & 15;
  int kc = l >> 2, part = l & 3;
  int lane2 = m + ((part & 1) << 4);
  int half  = part >> 1;
  size_t off = (((size_t)((j * 512 + rb) * 8 + kc) * 32) + lane2) * 16 + half * 8;
  *(v4u*)(pnA + off) = q;
}

// ---------------------------------------------------------------------------
// k2: fused per-head SwiGLU FFN with residual.
// grid (128, 16): blockIdx.x = 64-row block, blockIdx.y = head.
// 8 waves = 4 row-groups x 2 col-groups. cg0 -> gate(w1), cg1 -> up(w3).
// sched_group_barrier pins a load-ahead pipeline: B fragments are issued
// many instructions before the WMMAs that consume them.
// ---------------------------------------------------------------------------
__global__ __launch_bounds__(256) void k2_ffn(
    const unsigned short* __restrict__ pnA,
    const unsigned short* __restrict__ w1B, const unsigned short* __restrict__ w3B,
    const unsigned short* __restrict__ w2B,
    const float* __restrict__ Pbuf, float* __restrict__ out) {
  __shared__ float Ulds[4][32][16];   // up exchange, lane-linear
  __shared__ float Hlds[4][16][32];   // h natural [row][col] per row-group

  int n   = blockIdx.y;
  int rbx = blockIdx.x;
  int w   = threadIdx.x >> 5;
  int l   = threadIdx.x & 31;
  int rg  = w >> 1;
  int cg  = w & 1;
  int rb16  = rbx * 4 + rg;
  int rbase = rb16 * 16;

  const v16bf* pA = (const v16bf*)pnA;
  const v16bf* WB = cg ? (const v16bf*)w3B : (const v16bf*)w1B;
  const v16bf* W2 = (const v16bf*)w2B;

  // resident pn A fragments: 16 rows x K=256 (8 chunks of 16x32)
  v16bf A[8];
  size_t abase = ((size_t)(n * 512 + rb16)) * 256 + l;
#pragma unroll
  for (int kc = 0; kc < 8; kc++) A[kc] = pA[abase + kc * 32];

  const v8f zero = {0.f, 0.f, 0.f, 0.f, 0.f, 0.f, 0.f, 0.f};
  v8f qacc[8];
#pragma unroll
  for (int t = 0; t < 8; t++) qacc[t] = zero;

  int hrow = (l >= 16) ? 8 : 0;
  int hcol = l & 15;
  int arow = l & 15;
  int acb  = (l >= 16) ? 8 : 0;

  for (int hc = 0; hc < 16; hc++) {
    // ---- GEMM1: stage 16 B fragments, pipeline loads under WMMAs ----
    size_t wb = ((size_t)(n * 32 + hc * 2)) * 256 + l;
    v16bf b0[8], b1[8];
#pragma unroll
    for (int kc = 0; kc < 8; kc++) b0[kc] = WB[wb + kc * 32];
#pragma unroll
    for (int kc = 0; kc < 8; kc++) b1[kc] = WB[wb + 256 + kc * 32];

    v8f acc0 = zero, acc1 = zero;
#pragma unroll
    for (int kc = 0; kc < 8; kc++)
      acc0 = __builtin_amdgcn_wmma_f32_16x16x32_bf16(false, A[kc], false, b0[kc],
                                                     (short)0, acc0, false, false);
#pragma unroll
    for (int kc = 0; kc < 8; kc++)
      acc1 = __builtin_amdgcn_wmma_f32_16x16x32_bf16(false, A[kc], false, b1[kc],
                                                     (short)0, acc1, false, false);

    // desired pipeline: 16 vmem-loads ahead, then alternate wmma/load groups
    __builtin_amdgcn_sched_group_barrier(0x020, 16, 0); // 16 VMEM reads (8 frags)
    __builtin_amdgcn_sched_group_barrier(0x008, 4, 0);  // 4 WMMA
    __builtin_amdgcn_sched_group_barrier(0x020, 8, 0);  // 8 VMEM reads
    __builtin_amdgcn_sched_group_barrier(0x008, 4, 0);  // 4 WMMA
    __builtin_amdgcn_sched_group_barrier(0x020, 8, 0);  // 8 VMEM reads
    __builtin_amdgcn_sched_group_barrier(0x008, 8, 0);  // 8 WMMA

    if (cg) {  // up-wave publishes its C fragments lane-linear (4x ds_store_b128)
      float* up = &Ulds[rg][l][0];
      *(v4f*)(up + 0)  = __builtin_shufflevector(acc0, acc0, 0, 1, 2, 3);
      *(v4f*)(up + 4)  = __builtin_shufflevector(acc0, acc0, 4, 5, 6, 7);
      *(v4f*)(up + 8)  = __builtin_shufflevector(acc1, acc1, 0, 1, 2, 3);
      *(v4f*)(up + 12) = __builtin_shufflevector(acc1, acc1, 4, 5, 6, 7);
    }
    __syncthreads();

    if (!cg) {  // gate-wave: h = silu(g) * u, store naturally (transpose via LDS)
      const float* up = &Ulds[rg][l][0];
      v4f uu[4];
#pragma unroll
      for (int qq = 0; qq < 4; qq++) uu[qq] = *(const v4f*)(up + 4 * qq);
#pragma unroll
      for (int i = 0; i < 4; i++) {
        float g00 = acc0[i],     g01 = acc0[4 + i];
        float g10 = acc1[i],     g11 = acc1[4 + i];
        Hlds[rg][hrow + i][hcol]          = (g00 / (1.0f + __expf(-g00))) * uu[0][i];
        Hlds[rg][hrow + 4 + i][hcol]      = (g01 / (1.0f + __expf(-g01))) * uu[1][i];
        Hlds[rg][hrow + i][16 + hcol]     = (g10 / (1.0f + __expf(-g10))) * uu[2][i];
        Hlds[rg][hrow + 4 + i][16 + hcol] = (g11 / (1.0f + __expf(-g11))) * uu[3][i];
      }
    }
    __syncthreads();

    // ---- re-read h as bf16 A operand (16 rows x K=32), 4x ds_load_b128 ----
    const float* hp = &Hlds[rg][arow][0];
    v4f h0a = *(const v4f*)(hp + acb);
    v4f h0b = *(const v4f*)(hp + acb + 4);
    v4f h1a = *(const v4f*)(hp + 16 + acb);
    v4f h1b = *(const v4f*)(hp + 16 + acb + 4);
    v16u hs;
#pragma unroll
    for (int i = 0; i < 4; i++) {
      hs[i]      = f2bf(h0a[i]);
      hs[4 + i]  = f2bf(h0b[i]);
      hs[8 + i]  = f2bf(h1a[i]);
      hs[12 + i] = f2bf(h1b[i]);
    }
    v16bf ha = __builtin_bit_cast(v16bf, hs);

    // ---- GEMM2: stage 8 B fragments, then burst 8 WMMAs ----
    size_t w2b = ((size_t)(n * 256 + hc)) * 32 + l + (size_t)cg * 8 * 512;
    v16bf b2[8];
#pragma unroll
    for (int t = 0; t < 8; t++) b2[t] = W2[w2b + (size_t)t * 512];
#pragma unroll
    for (int t = 0; t < 8; t++)
      qacc[t] = __builtin_amdgcn_wmma_f32_16x16x32_bf16(false, ha, false, b2[t],
                                                        (short)0, qacc[t], false, false);

    __builtin_amdgcn_sched_group_barrier(0x020, 16, 0); // all 16 VMEM reads first
    __builtin_amdgcn_sched_group_barrier(0x008, 8, 0);  // then 8 WMMA
  }

  // ---- epilogue: q += P, write out ----
#pragma unroll
  for (int t = 0; t < 8; t++) {
    int col = (cg * 8 + t) * 16 + (l & 15);
#pragma unroll
    for (int rr = 0; rr < 8; rr++) {
      int rowb = rbase + rr + ((l >= 16) ? 8 : 0);
      size_t idx = (((size_t)rowb * 16) + n) * 256 + col;
      out[idx] = qacc[t][rr] + Pbuf[idx];
    }
  }
}

// ---------------------------------------------------------------------------
extern "C" void kernel_launch(void* const* d_in, const int* in_sizes, int n_in,
                              void* d_out, int out_size, void* d_ws, size_t ws_size,
                              hipStream_t stream) {
  const float* x   = (const float*)d_in[0];
  const float* n1w = (const float*)d_in[1];
  const float* n2w = (const float*)d_in[2];
  const float* w1  = (const float*)d_in[3];
  const float* w3  = (const float*)d_in[4];
  const float* w2  = (const float*)d_in[5];
  float* out = (float*)d_out;

  char* ws = (char*)d_ws;
  float*          Pbuf = (float*)ws;                                   // 128 MiB
  unsigned short* pnA  = (unsigned short*)(ws + (size_t)134217728);    //  64 MiB
  unsigned short* w1B  = (unsigned short*)(ws + (size_t)201326592);    //   4 MiB
  unsigned short* w3B  = (unsigned short*)(ws + (size_t)205520896);    //   4 MiB
  unsigned short* w2B  = (unsigned short*)(ws + (size_t)209715200);    //   4 MiB

  k0_wswz<<<1536, 256, 0, stream>>>(w1, w3, w2, w1B, w3B, w2B);
  k1_mix<<<B_, 512, 0, stream>>>(x, n1w, n2w, Pbuf, pnA);
  k2_ffn<<<dim3(128, 16), 256, 0, stream>>>(pnA, w1B, w3B, w2B, Pbuf, out);
}
